// LiquidAIModel_47193100648724
// MI455X (gfx1250) — compile-verified, matrix operand order checked by reading, not run on hardware
//
#include <hip/hip_runtime.h>

// ---------------------------------------------------------------------------
// MI455X (gfx1250) implementation.
// Everything heavy is a 128x128x32-tiled bf16 WMMA GEMM (f32 accumulate):
//   - fp32 tiles loaded from global, converted to bf16 while staged into LDS
//   - ping-pong LDS buffers: next tile's global loads overlap the WMMA block,
//     one s_barrier per K-step
//   - per-wave 32x64 sub-tile -> 8x v_wmma_f32_16x16x32_bf16 per K-step
//   - epilogue fully templated (bias/ReLU/residual/accumulate) -> straight-line
// ---------------------------------------------------------------------------

#define BM 128
#define BN 128
#define BK 32
#define BKP 40   // +8 bf16 pad: rows stay 16B aligned, staggers LDS banks

typedef __attribute__((ext_vector_type(16))) __bf16 bf16x16;
typedef __attribute__((ext_vector_type(8)))  __bf16 bf16x8;
typedef __attribute__((ext_vector_type(8)))  float  f32x8;

union FragU { bf16x16 v; bf16x8 h[2]; };

template<int TRANSB, int HAS_BIAS, int HAS_RESID, int ACCUM, int RELU>
__global__ __launch_bounds__(256)
void gemm_wmma_bf16(const float* __restrict__ A, int lda,
                    const float* __restrict__ B, int ldb,
                    const float* __restrict__ bias,
                    const float* __restrict__ resid, int ldr,
                    float* __restrict__ C, int ldc,
                    int M, int N, int K,
                    float alpha,
                    const float* __restrict__ alpha_dev,
                    const int*   __restrict__ sel,
                    long selB_stride, long selBias_stride)
{
    __shared__ __bf16 As[2][BM][BKP];
    __shared__ __bf16 Bs[2][BN][BKP];

    if (sel) {                               // MoE: expert chosen on device
        const int e = sel[0];
        B += (long)e * selB_stride;
        if (HAS_BIAS) bias += (long)e * selBias_stride;
    }
    const float aeff = alpha * (alpha_dev ? alpha_dev[0] : 1.0f);

    const int tid  = threadIdx.x;
    const int lane = tid & 31;
    const int lrow = lane & 15;              // M (A frag) / N (B frag) index
    const int hsel = lane >> 4;              // K-chunk select per ISA layout
    const int wid  = tid >> 5;
    const int wm   = (wid & 3) * 32;         // 4 wave-rows x 2 wave-cols
    const int wn   = (wid >> 2) * 64;
    const int m0   = blockIdx.y * BM;
    const int n0   = blockIdx.x * BN;

    // ---- staging helpers (uniform addressing, coalesced) ----
    auto loadA = [&](int k0, float4* ra) {
        #pragma unroll
        for (int i = 0; i < 4; ++i) {
            const int f = tid + i * 256;     // 1024 float4 = 128x32 f32
            const int r = f >> 3;
            const int c = (f & 7) * 4;
            ra[i] = *(const float4*)(A + (size_t)(m0 + r) * lda + (k0 + c));
        }
    };
    auto storeA = [&](int buf, const float4* ra) {
        #pragma unroll
        for (int i = 0; i < 4; ++i) {
            const int f = tid + i * 256;
            __bf16* d = &As[buf][f >> 3][(f & 7) * 4];
            d[0] = (__bf16)ra[i].x; d[1] = (__bf16)ra[i].y;
            d[2] = (__bf16)ra[i].z; d[3] = (__bf16)ra[i].w;
        }
    };
    auto loadB = [&](int k0, float4* rb4, float* rbs) {
        if (TRANSB) {                        // B is [N,K] row-major (weights)
            #pragma unroll
            for (int i = 0; i < 4; ++i) {
                const int f = tid + i * 256;
                const int r = f >> 3;
                const int c = (f & 7) * 4;
                rb4[i] = *(const float4*)(B + (size_t)(n0 + r) * ldb + (k0 + c));
            }
        } else {                             // B is [K,N] row-major (probs@V)
            #pragma unroll
            for (int i = 0; i < 16; ++i) {
                const int e  = tid + i * 256;   // coalesced along N
                const int kk = e >> 7;
                const int nn = e & 127;
                rbs[i] = B[(size_t)(k0 + kk) * ldb + (n0 + nn)];
            }
        }
    };
    auto storeB = [&](int buf, const float4* rb4, const float* rbs) {
        if (TRANSB) {
            #pragma unroll
            for (int i = 0; i < 4; ++i) {
                const int f = tid + i * 256;
                __bf16* d = &Bs[buf][f >> 3][(f & 7) * 4];
                d[0] = (__bf16)rb4[i].x; d[1] = (__bf16)rb4[i].y;
                d[2] = (__bf16)rb4[i].z; d[3] = (__bf16)rb4[i].w;
            }
        } else {
            #pragma unroll
            for (int i = 0; i < 16; ++i) {
                const int e = tid + i * 256;
                Bs[buf][e & 127][e >> 7] = (__bf16)rbs[i];
            }
        }
    };

    f32x8 acc[2][4];
    #pragma unroll
    for (int i = 0; i < 2; ++i)
        #pragma unroll
        for (int j = 0; j < 4; ++j)
            #pragma unroll
            for (int q = 0; q < 8; ++q) acc[i][j][q] = 0.0f;

    // ---- software pipeline: prologue stages tile 0 ----
    const int nK = K / BK;
    float4 ra[4], rb4[4]; float rbs[16];
    loadA(0, ra);  loadB(0, rb4, rbs);
    storeA(0, ra); storeB(0, rb4, rbs);
    __syncthreads();

    for (int kt = 0; kt < nK; ++kt) {
        const int  cur  = kt & 1;
        const bool more = (kt + 1 < nK);
        if (more) {                          // issue next tile's global loads
            loadA((kt + 1) * BK, ra);        // (overlaps WMMA below)
            loadB((kt + 1) * BK, rb4, rbs);
        }

        // ---- fragment loads per ISA 16-bit A/B layout ----
        FragU af[2], bfr[4];
        #pragma unroll
        for (int mi = 0; mi < 2; ++mi) {
            const __bf16* rp = &As[cur][wm + mi * 16 + lrow][0];
            af[mi].h[0] = *(const bf16x8*)(rp + hsel * 8);        // K 0-7 / 8-15
            af[mi].h[1] = *(const bf16x8*)(rp + 16 + hsel * 8);   // K 16-23 / 24-31
        }
        #pragma unroll
        for (int ni = 0; ni < 4; ++ni) {
            const __bf16* rp = &Bs[cur][wn + ni * 16 + lrow][0];
            bfr[ni].h[0] = *(const bf16x8*)(rp + hsel * 8);
            bfr[ni].h[1] = *(const bf16x8*)(rp + 16 + hsel * 8);
        }
        #pragma unroll
        for (int mi = 0; mi < 2; ++mi)
            #pragma unroll
            for (int ni = 0; ni < 4; ++ni)
                acc[mi][ni] = __builtin_amdgcn_wmma_f32_16x16x32_bf16(
                    false, af[mi].v, false, bfr[ni].v,
                    (short)0, acc[mi][ni], false, false);

        if (more) {                          // write next tile into alt buffer
            storeA(1 - cur, ra);
            storeB(1 - cur, rb4, rbs);
        }
        __syncthreads();                     // one barrier per K-step
    }

    // ---- straight-line fused epilogue ----
    #pragma unroll
    for (int mi = 0; mi < 2; ++mi) {
        #pragma unroll
        for (int ni = 0; ni < 4; ++ni) {
            const int cg = n0 + wn + ni * 16 + lrow;
            const float bv = HAS_BIAS ? bias[cg] : 0.0f;
            #pragma unroll
            for (int j = 0; j < 8; ++j) {
                const int rg = m0 + wm + mi * 16 + hsel * 8 + j;
                float v = (acc[mi][ni][j] + bv) * aeff;
                if (RELU)      v = v > 0.0f ? v : 0.0f;
                if (HAS_RESID) v += resid[(size_t)rg * ldr + cg];
                float* cp = C + (size_t)rg * ldc + cg;
                if (ACCUM)     v += *cp;
                *cp = v;
            }
        }
    }
}

// ---- token-0 gate: logits, softmax(4), top-2 (stable: lower idx on ties) ----
__global__ __launch_bounds__(256)
void gate_kernel(const float* __restrict__ X, const float* __restrict__ wg,
                 const float* __restrict__ bg, int din,
                 int* __restrict__ sel, float* __restrict__ sc)
{
    __shared__ float red[4][256];
    const int tid = threadIdx.x;
    float a0 = 0.f, a1 = 0.f, a2 = 0.f, a3 = 0.f;
    for (int i = tid; i < din; i += 256) {
        const float xv = X[i];
        a0 += xv * wg[i];
        a1 += xv * wg[din + i];
        a2 += xv * wg[2 * din + i];
        a3 += xv * wg[3 * din + i];
    }
    red[0][tid] = a0; red[1][tid] = a1; red[2][tid] = a2; red[3][tid] = a3;
    __syncthreads();
    for (int o = 128; o > 0; o >>= 1) {
        if (tid < o) {
            red[0][tid] += red[0][tid + o]; red[1][tid] += red[1][tid + o];
            red[2][tid] += red[2][tid + o]; red[3][tid] += red[3][tid + o];
        }
        __syncthreads();
    }
    if (tid == 0) {
        float l[4], p[4], mx = -1e30f, s = 0.f;
        for (int e = 0; e < 4; ++e) { l[e] = red[e][0] + bg[e]; mx = fmaxf(mx, l[e]); }
        for (int e = 0; e < 4; ++e) { p[e] = __expf(l[e] - mx); s += p[e]; }
        for (int e = 0; e < 4; ++e) p[e] /= s;
        int i0 = 0;
        for (int e = 1; e < 4; ++e) if (p[e] > p[i0]) i0 = e;
        int i1 = (i0 == 0) ? 1 : 0;
        for (int e = 0; e < 4; ++e) if (e != i0 && p[e] > p[i1]) i1 = e;
        sel[0] = i0; sel[1] = i1; sc[0] = p[i0]; sc[1] = p[i1];
    }
}

// ---- in-place row softmax (scale already applied by GEMM alpha) ----
__global__ __launch_bounds__(256)
void softmax_rows(float* __restrict__ X, int W)
{
    float* x = X + (size_t)blockIdx.x * W;
    const int tid = threadIdx.x;
    __shared__ float red[256];
    float mx = -1e30f;
    for (int i = tid; i < W; i += 256) mx = fmaxf(mx, x[i]);
    red[tid] = mx; __syncthreads();
    for (int o = 128; o > 0; o >>= 1) {
        if (tid < o) red[tid] = fmaxf(red[tid], red[tid + o]);
        __syncthreads();
    }
    mx = red[0]; __syncthreads();
    float s = 0.f;
    for (int i = tid; i < W; i += 256) { const float e = __expf(x[i] - mx); x[i] = e; s += e; }
    red[tid] = s; __syncthreads();
    for (int o = 128; o > 0; o >>= 1) {
        if (tid < o) red[tid] += red[tid + o];
        __syncthreads();
    }
    const float inv = 1.0f / red[0];
    for (int i = tid; i < W; i += 256) x[i] *= inv;
}

// ---- LayerNorm, one block per row (in-place safe) ----
__global__ __launch_bounds__(256)
void ln_kernel(const float* __restrict__ X, const float* __restrict__ g,
               const float* __restrict__ b, float* __restrict__ Y, int W)
{
    const float* x = X + (size_t)blockIdx.x * W;
    float*       y = Y + (size_t)blockIdx.x * W;
    const int tid = threadIdx.x;
    __shared__ float rs[256], rss[256];
    float s = 0.f, ss = 0.f;
    for (int i = tid; i < W; i += 256) { const float v = x[i]; s += v; ss += v * v; }
    rs[tid] = s; rss[tid] = ss;
    __syncthreads();
    for (int o = 128; o > 0; o >>= 1) {
        if (tid < o) { rs[tid] += rs[tid + o]; rss[tid] += rss[tid + o]; }
        __syncthreads();
    }
    const float mean = rs[0] / W;
    const float var  = rss[0] / W - mean * mean;
    const float inv  = rsqrtf(var + 1e-5f);
    for (int i = tid; i < W; i += 256)
        y[i] = (x[i] - mean) * inv * g[i] + b[i];
}

// ---------------------------------------------------------------------------
// Host-side orchestration
// ---------------------------------------------------------------------------
static inline void gemm(hipStream_t st,
                        const float* A, int lda, const float* B, int ldb,
                        const float* bias, const float* resid, int ldr,
                        float* C, int ldc, int M, int N, int K,
                        float alpha = 1.0f, const float* alpha_dev = nullptr,
                        const int* sel = nullptr, long selB = 0, long selBias = 0,
                        int accum = 0, int relu = 0, int transB = 1)
{
    dim3 grid(N / BN, M / BM), blk(256);
    #define LAUNCH(TB, HB, HR, AC, RL)                                         \
        gemm_wmma_bf16<TB, HB, HR, AC, RL><<<grid, blk, 0, st>>>(              \
            A, lda, B, ldb, bias, resid, ldr, C, ldc, M, N, K,                 \
            alpha, alpha_dev, sel, selB, selBias)
    if (!transB)            LAUNCH(0, 0, 0, 0, 0);   // probs @ V
    else if (relu)          LAUNCH(1, 1, 0, 0, 1);   // ff1
    else if (resid)         LAUNCH(1, 1, 1, 0, 0);   // wo / ff2
    else if (accum)         LAUNCH(1, 1, 0, 1, 0);   // MoE expert #2
    else if (bias)          LAUNCH(1, 1, 0, 0, 0);   // projections / MoE #1
    else                    LAUNCH(1, 0, 0, 0, 0);   // Q @ K^T
    #undef LAUNCH
}

extern "C" void kernel_launch(void* const* d_in, const int* in_sizes, int n_in,
                              void* d_out, int out_size, void* d_ws, size_t ws_size,
                              hipStream_t stream)
{
    (void)in_sizes; (void)n_in; (void)out_size; (void)ws_size;
    const int S = 2048, D = 1024, Hd = 1024, FF = 2048, NHEAD = 4, dh = 256;
    auto F = [&](int i) { return (const float*)d_in[i]; };

    // Workspace layout (fp32): ~128 MB total.
    char* w = (char*)d_ws;
    const size_t SH = (size_t)S * Hd * sizeof(float);      // 8 MB
    float* bufT  = (float*)(w);                            // t / h
    float* bufM  = (float*)(w + 1 * SH);                   // mem
    float* bufQ  = (float*)(w + 2 * SH);
    float* bufK  = (float*)(w + 3 * SH);
    float* bufV  = (float*)(w + 4 * SH);
    float* bufA  = (float*)(w + 5 * SH);                   // attn concat out
    float* bufFF = (float*)(w + 6 * SH);                   // S x FF (2*SH)
    float* bufSC = (float*)(w + 8 * SH);                   // 4 x S x S (8*SH)
    float* gsc   = (float*)(w + 16 * SH);                  // gate scores [2]
    int*   gsel  = (int*)  (w + 16 * SH + 2 * sizeof(float)); // gate idx [2]

    // ---- MoE stack (token-0 routing applied to all tokens) ----
    auto moe_layer = [&](const float* in, float* out, int base, int din) {
        gate_kernel<<<1, 256, 0, stream>>>(in, F(base + 0), F(base + 1), din, gsel, gsc);
        gemm(stream, in, din, F(base + 2), din, F(base + 3), nullptr, 0, out, Hd,
             S, Hd, din, 1.0f, gsc + 0, gsel + 0, (long)Hd * din, (long)Hd, 0, 0, 1);
        gemm(stream, in, din, F(base + 2), din, F(base + 3), nullptr, 0, out, Hd,
             S, Hd, din, 1.0f, gsc + 1, gsel + 1, (long)Hd * din, (long)Hd, 1, 0, 1);
    };
    moe_layer((const float*)d_in[0], bufT, 1, D);
    moe_layer(bufT, bufM, 5, Hd);
    moe_layer(bufM, bufT, 9, Hd);                          // h -> bufT
    hipMemcpyAsync(bufM, bufT, SH, hipMemcpyDeviceToDevice, stream);  // mem = h

    // ---- multi-head attention: out = resid + (attn(xq,xkv) @ wo^T + bo) ----
    auto mha = [&](const float* xq, const float* xkv, int pb,
                   const float* residp, float* outp) {
        gemm(stream, xq,  Hd, F(pb + 0), Hd, F(pb + 1), nullptr, 0, bufQ, Hd, S, Hd, Hd);
        gemm(stream, xkv, Hd, F(pb + 2), Hd, F(pb + 3), nullptr, 0, bufK, Hd, S, Hd, Hd);
        gemm(stream, xkv, Hd, F(pb + 4), Hd, F(pb + 5), nullptr, 0, bufV, Hd, S, Hd, Hd);
        for (int h = 0; h < NHEAD; ++h)                    // Qh @ Kh^T / sqrt(dh)
            gemm(stream, bufQ + h * dh, Hd, bufK + h * dh, Hd, nullptr, nullptr, 0,
                 bufSC + (size_t)h * S * S, S, S, S, dh, 0.0625f);
        softmax_rows<<<NHEAD * S, 256, 0, stream>>>(bufSC, S);
        for (int h = 0; h < NHEAD; ++h)                    // probs @ Vh  (B not transposed)
            gemm(stream, bufSC + (size_t)h * S * S, S, bufV + h * dh, Hd,
                 nullptr, nullptr, 0, bufA + h * dh, Hd, S, dh, S,
                 1.0f, nullptr, nullptr, 0, 0, 0, 0, /*transB=*/0);
        gemm(stream, bufA, Hd, F(pb + 6), Hd, F(pb + 7), residp, Hd, outp, Hd, S, Hd, Hd);
    };

    // ---- encoder (post-norm, ReLU FFN) ----
    for (int j = 0; j < 2; ++j) {
        const int b = 13 + 16 * j;
        mha(bufM, bufM, b, bufM, bufQ);
        ln_kernel<<<S, 256, 0, stream>>>(bufQ, F(b + 8), F(b + 9), bufM, Hd);
        gemm(stream, bufM, Hd, F(b + 10), Hd, F(b + 11), nullptr, 0, bufFF, FF,
             S, FF, Hd, 1.0f, nullptr, nullptr, 0, 0, 0, /*relu=*/1, 1);
        gemm(stream, bufFF, FF, F(b + 12), FF, F(b + 13), bufM, Hd, bufQ, Hd, S, Hd, FF);
        ln_kernel<<<S, 256, 0, stream>>>(bufQ, F(b + 14), F(b + 15), bufM, Hd);
    }
    ln_kernel<<<S, 256, 0, stream>>>(bufM, F(45), F(46), bufM, Hd);   // enc_norm

    // ---- decoder (self-attn, cross-attn, FFN) ----
    for (int j = 0; j < 2; ++j) {
        const int b = 47 + 26 * j;
        mha(bufT, bufT, b + 0, bufT, bufQ);
        ln_kernel<<<S, 256, 0, stream>>>(bufQ, F(b + 8),  F(b + 9),  bufT, Hd);
        mha(bufT, bufM, b + 10, bufT, bufQ);
        ln_kernel<<<S, 256, 0, stream>>>(bufQ, F(b + 18), F(b + 19), bufT, Hd);
        gemm(stream, bufT, Hd, F(b + 20), Hd, F(b + 21), nullptr, 0, bufFF, FF,
             S, FF, Hd, 1.0f, nullptr, nullptr, 0, 0, 0, /*relu=*/1, 1);
        gemm(stream, bufFF, FF, F(b + 22), FF, F(b + 23), bufT, Hd, bufQ, Hd, S, Hd, FF);
        ln_kernel<<<S, 256, 0, stream>>>(bufQ, F(b + 24), F(b + 25), bufT, Hd);
    }
    ln_kernel<<<S, 256, 0, stream>>>(bufT, F(99), F(100), bufT, Hd);  // dec_norm

    // ---- output head ----
    gemm(stream, bufT, Hd, F(101), Hd, F(102), nullptr, 0, bufQ, Hd, S, Hd, Hd);
    gemm(stream, bufQ, Hd, F(103), Hd, F(104), nullptr, 0, (float*)d_out, 1024,
         S, 1024, Hd);
}
